// MIL_15178414424097
// MI455X (gfx1250) — matrix operand order repaired and checked by reading, not compiled, not run on hardware
//
#include <hip/hip_runtime.h>
#include <hip/hip_bf16.h>
#include <math.h>

// ---------------------------------------------------------------------------
// Problem constants (from reference)
// ---------------------------------------------------------------------------
#define NN      10000      // N_NODES
#define NE      160000     // N_EDGES
#define INDIM   1024
#define HID     512
#define CATW    2048       // HID * 4
#define OUTD    4
#define EPSMSG  1e-7f

// WMMA GEMM tiling
#define GT 64              // output tile (rows x cols) per 128-thread block
#define GK 32              // K step (matches 16x16x32 bf16 WMMA)
#define LPAD 16            // LDS row pad: stride 48 elems = 96B, keeps 16B align

// activations for GEMM epilogue
#define ACT_NONE   0
#define ACT_RELU   1
#define ACT_TANH   2
#define ACT_SIG    3
#define ACT_SIGMUL 4       // C *= sigmoid(acc + bias)  (attention gate fuse)

typedef __attribute__((ext_vector_type(16))) __bf16 v16bf;
typedef __attribute__((ext_vector_type(8)))  float  v8f;

union FragU { uint4 q[2]; v16bf bf; };

// f32 -> bf16 (round to nearest even), returning raw bits
__device__ __forceinline__ unsigned short bfbits(float f) {
    unsigned u = __float_as_uint(f);
    return (unsigned short)((u + 0x7FFFu + ((u >> 16) & 1u)) >> 16);
}
__device__ __forceinline__ unsigned bfpack2(float lo, float hi) {
    return (unsigned)bfbits(lo) | ((unsigned)bfbits(hi) << 16);
}

// float atomic max via signed/unsigned integer monotonicity trick
__device__ __forceinline__ void atomicMaxF32(float* addr, float val) {
    if (val >= 0.0f) atomicMax((int*)addr, __float_as_int(val));
    else             atomicMin((unsigned int*)addr, __float_as_uint(val));
}

// async 16B global->LDS copy (GLOBAL_LOAD_ASYNC_TO_LDS_B128, tracked by ASYNCcnt).
// LDS address = low 32 bits of the generic pointer (ISA §10.2 aperture rule).
__device__ __forceinline__ void async_copy16(const unsigned short* g, unsigned short* lds) {
    unsigned ldsAddr = (unsigned)(uintptr_t)lds;
    unsigned long long gAddr = (unsigned long long)(uintptr_t)g;
    asm volatile("global_load_async_to_lds_b128 %0, %1, off"
                 :: "v"(ldsAddr), "v"(gAddr) : "memory");
}
__device__ __forceinline__ void wait_async0() {
#if __has_builtin(__builtin_amdgcn_s_wait_asynccnt)
    __builtin_amdgcn_s_wait_asynccnt(0);
#else
    asm volatile("s_wait_asynccnt 0x0" ::: "memory");
#endif
}

// ---------------------------------------------------------------------------
// bf16 pre-conversion kernels (run once per operand, not per tile)
// ---------------------------------------------------------------------------
__global__ void cvt_bf16_kernel(const float* __restrict__ in,
                                unsigned short* __restrict__ out, long n)
{
    long i = ((long)blockIdx.x * blockDim.x + threadIdx.x) * 4;
    if (i >= n) return;
    float4 v = *(const float4*)(in + i);
    uint2 u; u.x = bfpack2(v.x, v.y); u.y = bfpack2(v.z, v.w);
    *(uint2*)(out + i) = u;
    __builtin_prefetch(in + i + 4096, 0, 1);   // global_prefetch_b8
}

// W[K,M] f32  ->  Wt[M,K] bf16 (transpose-convert, 32x32 LDS tiles)
__global__ __launch_bounds__(256) void transpose_cvt_kernel(
    const float* __restrict__ W, unsigned short* __restrict__ Wt, int K, int M)
{
    __shared__ float tile[32][33];
    int tx = threadIdx.x, ty = threadIdx.y;          // 32 x 8
    int mb = blockIdx.x * 32, kb = blockIdx.y * 32;
    #pragma unroll
    for (int j = 0; j < 4; ++j)
        tile[ty + 8 * j][tx] = W[(long)(kb + ty + 8 * j) * M + mb + tx];
    __syncthreads();
    #pragma unroll
    for (int j = 0; j < 4; ++j)
        Wt[(long)(mb + ty + 8 * j) * K + kb + tx] = bfbits(tile[tx][ty + 8 * j]);
}

// ---------------------------------------------------------------------------
// WMMA GEMM:  C[nrows,M] = act(A[nrows,K] @ B[K,M] + bias)
//   A:  bf16 row-major [nrows,K] pitch lda
//   Bt: bf16 row-major [M,K] (pre-transposed weights) pitch ldbt
//   Requires K % 32 == 0, M % 64 == 0 (all call sites: 512/1024/2048).
//   128 threads = 4 waves; each wave owns a 32x32 quadrant = 2x2 WMMA tiles.
//   Staging uses async global->LDS b128 copies; fragments are pure b128 LDS
//   loads thanks to the [*,K-contiguous] layouts matching ISA §7.12.2.
// ---------------------------------------------------------------------------
__global__ __launch_bounds__(128) void wmma_gemm_kernel(
    const unsigned short* __restrict__ A, int lda,
    const unsigned short* __restrict__ Bt, int ldbt,
    const float* __restrict__ bias,
    float* __restrict__ C, int ldc,
    int nrows, int K, int act)
{
    __shared__ unsigned short As[GT][GK + LPAD];   // [row][k]
    __shared__ unsigned short Bs[GT][GK + LPAD];   // [col][k]  (B transposed)

    const int tid  = threadIdx.x;
    const int lane = tid & 31;
    const int wave = tid >> 5;
    const int wr   = (wave >> 1) * 32;
    const int wc   = (wave & 1) * 32;
    const int rowBlk = blockIdx.y * GT;
    const int colBlk = blockIdx.x * GT;
    const int half = lane >> 4;
    const int l15  = lane & 15;

    v8f acc00 = {0,0,0,0,0,0,0,0};
    v8f acc01 = {0,0,0,0,0,0,0,0};
    v8f acc10 = {0,0,0,0,0,0,0,0};
    v8f acc11 = {0,0,0,0,0,0,0,0};

    for (int k0 = 0; k0 < K; k0 += GK) {
        __syncthreads();
        // stage A (64 x 64B) and B (64 x 64B) tiles: 256 16B chunks each,
        // 2 per thread, via async copies (no VGPR round-trip, no conversion).
        #pragma unroll
        for (int i = 0; i < 2; ++i) {
            int idx = tid + i * 128;
            int r  = idx >> 2;
            int c8 = (idx & 3) * 8;
            int gr = rowBlk + r;
            if (gr < nrows) {
                async_copy16(A + (long)gr * lda + k0 + c8, &As[r][c8]);
            } else {
                *(uint4*)&As[r][c8] = make_uint4(0u, 0u, 0u, 0u);
            }
            async_copy16(Bt + (long)(colBlk + r) * ldbt + k0 + c8, &Bs[r][c8]);
        }
        wait_async0();
        __syncthreads();

        // fragments (ISA §7.12.2):
        //  A 16x32: lane half selects 8-wide K subblock -> elems = As[m][8h .. ] & [16+8h ..]
        //  B 32x16: elems = Bs[n][8h .. ] & [16+8h ..]
        FragU a0, a1, b0, b1;
        {
            const unsigned short* ra = &As[wr + l15][0];
            a0.q[0] = *(const uint4*)(ra + 8 * half);
            a0.q[1] = *(const uint4*)(ra + 16 + 8 * half);
            const unsigned short* rb = &As[wr + 16 + l15][0];
            a1.q[0] = *(const uint4*)(rb + 8 * half);
            a1.q[1] = *(const uint4*)(rb + 16 + 8 * half);
            const unsigned short* rc = &Bs[wc + l15][0];
            b0.q[0] = *(const uint4*)(rc + 8 * half);
            b0.q[1] = *(const uint4*)(rc + 16 + 8 * half);
            const unsigned short* rd = &Bs[wc + 16 + l15][0];
            b1.q[0] = *(const uint4*)(rd + 8 * half);
            b1.q[1] = *(const uint4*)(rd + 16 + 8 * half);
        }
        acc00 = __builtin_amdgcn_wmma_f32_16x16x32_bf16(false, a0.bf, false, b0.bf, (short)0, acc00, false, false);
        acc01 = __builtin_amdgcn_wmma_f32_16x16x32_bf16(false, a0.bf, false, b1.bf, (short)0, acc01, false, false);
        acc10 = __builtin_amdgcn_wmma_f32_16x16x32_bf16(false, a1.bf, false, b0.bf, (short)0, acc10, false, false);
        acc11 = __builtin_amdgcn_wmma_f32_16x16x32_bf16(false, a1.bf, false, b1.bf, (short)0, acc11, false, false);
    }

    // epilogue: C/D layout -> row = base + v + 8*half, col = base + (lane&15)
    const v8f* accs[4] = { &acc00, &acc01, &acc10, &acc11 };
    #pragma unroll
    for (int t = 0; t < 4; ++t) {
        int rBase = rowBlk + wr + ((t >> 1) ? 16 : 0);
        int cBase = colBlk + wc + ((t & 1) ? 16 : 0);
        const v8f a = *accs[t];
        #pragma unroll
        for (int v = 0; v < 8; ++v) {
            int r = rBase + v + 8 * half;
            if (r < nrows) {
                int c = cBase + l15;
                float lin = a[v] + bias[c];
                float* p = C + (long)r * ldc + c;
                if (act == ACT_RELU)        *p = fmaxf(lin, 0.f);
                else if (act == ACT_TANH)   *p = tanhf(lin);
                else if (act == ACT_SIG)    *p = 1.f / (1.f + __expf(-lin));
                else if (act == ACT_SIGMUL) *p = (*p) * (1.f / (1.f + __expf(-lin)));
                else                        *p = lin;
            }
        }
    }
}

// ---------------------------------------------------------------------------
// Elementwise / scatter kernels
// ---------------------------------------------------------------------------
__global__ void fill_kernel(float* __restrict__ p, float v, long n) {
    long i = (long)blockIdx.x * blockDim.x + threadIdx.x;
    if (i < n) p[i] = v;
}

// pass 1: per-(dst,feature) max of z = t * (relu(x[src]) + eps)
__global__ void edge_max_kernel(const float* __restrict__ xin, int pitch,
                                const int* __restrict__ src, const int* __restrict__ dst,
                                const float* __restrict__ tptr,
                                float* __restrict__ zmax)
{
    long gid = (long)blockIdx.x * blockDim.x + threadIdx.x;
    const long total = (long)NE * (HID / 4);
    if (gid >= total) return;
    int e = (int)(gid >> 7);          // HID/4 == 128
    int q = (int)(gid & 127) * 4;
    int s = src[e], d = dst[e];
    float t = tptr[0];
    float4 xv = *(const float4*)(xin + (long)s * pitch + q);
    float* zp = zmax + (long)d * HID + q;
    atomicMaxF32(zp + 0, t * (fmaxf(xv.x, 0.f) + EPSMSG));
    atomicMaxF32(zp + 1, t * (fmaxf(xv.y, 0.f) + EPSMSG));
    atomicMaxF32(zp + 2, t * (fmaxf(xv.z, 0.f) + EPSMSG));
    atomicMaxF32(zp + 3, t * (fmaxf(xv.w, 0.f) + EPSMSG));
}

// pass 2: denom += exp(z - zmax[dst]);  numer += msg * exp(z - zmax[dst])
__global__ void edge_sum_kernel(const float* __restrict__ xin, int pitch,
                                const int* __restrict__ src, const int* __restrict__ dst,
                                const float* __restrict__ tptr,
                                const float* __restrict__ zmax,
                                float* __restrict__ denom, float* __restrict__ numer)
{
    long gid = (long)blockIdx.x * blockDim.x + threadIdx.x;
    const long total = (long)NE * (HID / 4);
    if (gid >= total) return;
    int e = (int)(gid >> 7);
    int q = (int)(gid & 127) * 4;
    int s = src[e], d = dst[e];
    float t = tptr[0];
    float4 xv = *(const float4*)(xin + (long)s * pitch + q);
    long off = (long)d * HID + q;
    float m[4] = { fmaxf(xv.x, 0.f) + EPSMSG, fmaxf(xv.y, 0.f) + EPSMSG,
                   fmaxf(xv.z, 0.f) + EPSMSG, fmaxf(xv.w, 0.f) + EPSMSG };
    #pragma unroll
    for (int i = 0; i < 4; ++i) {
        float ez = __expf(t * m[i] - zmax[off + i]);
        atomicAdd(denom + off + i, ez);
        atomicAdd(numer + off + i, m[i] * ez);
    }
}

// finalize: out_node = numer / (denom + 1e-16) + x   (segment-softmax agg + root)
__global__ void node_out_kernel(const float* __restrict__ xin, int pitch,
                                const float* __restrict__ numer,
                                const float* __restrict__ denom,
                                float* __restrict__ out_node)
{
    long gid = (long)blockIdx.x * blockDim.x + threadIdx.x;
    if (gid >= (long)NN * HID) return;
    int n = (int)(gid / HID);
    int f = (int)(gid % HID);
    out_node[gid] = numer[gid] / (denom[gid] + 1e-16f) + xin[(long)n * pitch + f];
}

// row LayerNorm, optional relu, optional residual-add; pitched in/out/res
__global__ __launch_bounds__(256) void ln_kernel(
    const float* __restrict__ in, int in_pitch,
    float* __restrict__ out, int out_pitch,
    const float* __restrict__ g, const float* __restrict__ b,
    const float* __restrict__ res, int res_pitch,
    int dim, int do_relu)
{
    __shared__ float red[256];
    int row = blockIdx.x, tid = threadIdx.x;
    const float* ip = in + (long)row * in_pitch;
    float s = 0.f;
    for (int c = tid; c < dim; c += 256) s += ip[c];
    red[tid] = s; __syncthreads();
    for (int st = 128; st > 0; st >>= 1) { if (tid < st) red[tid] += red[tid + st]; __syncthreads(); }
    float mu = red[0] / dim; __syncthreads();
    float v = 0.f;
    for (int c = tid; c < dim; c += 256) { float d = ip[c] - mu; v += d * d; }
    red[tid] = v; __syncthreads();
    for (int st = 128; st > 0; st >>= 1) { if (tid < st) red[tid] += red[tid + st]; __syncthreads(); }
    float rstd = rsqrtf(red[0] / dim + 1e-5f);
    float* op = out + (long)row * out_pitch;
    const float* rp = res ? res + (long)row * res_pitch : nullptr;
    for (int c = tid; c < dim; c += 256) {
        float h = (ip[c] - mu) * rstd * g[c] + b[c];
        if (do_relu) h = fmaxf(h, 0.f);
        if (rp) h += rp[c];
        op[c] = h;
    }
}

// attention score per node: A[n] = sum_f ag[n,f] * cw[f] + cb
__global__ __launch_bounds__(256) void attn_score_kernel(
    const float* __restrict__ ag, const float* __restrict__ cw,
    const float* __restrict__ cb, float* __restrict__ scores)
{
    __shared__ float red[256];
    int n = blockIdx.x, tid = threadIdx.x;
    const float* row = ag + (long)n * CATW;
    float s = 0.f;
    for (int f = tid; f < CATW; f += 256) s += row[f] * cw[f];
    red[tid] = s; __syncthreads();
    for (int st = 128; st > 0; st >>= 1) { if (tid < st) red[tid] += red[tid + st]; __syncthreads(); }
    if (tid == 0) scores[n] = red[0] + cb[0];
}

// softmax over all nodes (single workgroup of 1024 threads)
__global__ __launch_bounds__(1024) void softmax_nodes_kernel(
    const float* __restrict__ scores, float* __restrict__ w, int n)
{
    __shared__ float red[1024];
    int tid = threadIdx.x;
    float m = -INFINITY;
    for (int i = tid; i < n; i += 1024) m = fmaxf(m, scores[i]);
    red[tid] = m; __syncthreads();
    for (int st = 512; st > 0; st >>= 1) { if (tid < st) red[tid] = fmaxf(red[tid], red[tid + st]); __syncthreads(); }
    float mx = red[0]; __syncthreads();
    float s = 0.f;
    for (int i = tid; i < n; i += 1024) s += __expf(scores[i] - mx);
    red[tid] = s; __syncthreads();
    for (int st = 512; st > 0; st >>= 1) { if (tid < st) red[tid] += red[tid + st]; __syncthreads(); }
    float inv = 1.f / red[0];
    for (int i = tid; i < n; i += 1024) w[i] = __expf(scores[i] - mx) * inv;
}

// pooled[f] += sum over node chunk of w[n]*hp[n,f]   (grid.y = node chunks)
__global__ __launch_bounds__(256) void pooled_kernel(
    const float* __restrict__ hp, const float* __restrict__ w,
    float* __restrict__ pooled, int chunk)
{
    int f = blockIdx.x * 256 + threadIdx.x;
    int n0 = blockIdx.y * chunk;
    int n1 = n0 + chunk; if (n1 > NN) n1 = NN;
    float s = 0.f;
    for (int n = n0; n < n1; ++n) s += w[n] * hp[(long)n * CATW + f];
    atomicAdd(pooled + f, s);
}

// r[m] = relu(sum_k pooled[k]*W[k,m] + b[m])
__global__ __launch_bounds__(256) void rho_kernel(
    const float* __restrict__ pooled, const float* __restrict__ W,
    const float* __restrict__ b, float* __restrict__ r)
{
    int m = blockIdx.x * 256 + threadIdx.x;
    float s = b[m];
    for (int k = 0; k < CATW; ++k) s += pooled[k] * W[(long)k * CATW + m];
    r[m] = fmaxf(s, 0.f);
}

// logits -> sigmoid, OUT=4
__global__ __launch_bounds__(256) void cls_kernel(
    const float* __restrict__ r, const float* __restrict__ W,
    const float* __restrict__ b, float* __restrict__ out)
{
    __shared__ float red[4][256];
    int tid = threadIdx.x;
    float s0 = 0.f, s1 = 0.f, s2 = 0.f, s3 = 0.f;
    for (int k = tid; k < CATW; k += 256) {
        float rv = r[k];
        s0 += rv * W[k * 4 + 0]; s1 += rv * W[k * 4 + 1];
        s2 += rv * W[k * 4 + 2]; s3 += rv * W[k * 4 + 3];
    }
    red[0][tid] = s0; red[1][tid] = s1; red[2][tid] = s2; red[3][tid] = s3;
    __syncthreads();
    for (int st = 128; st > 0; st >>= 1) {
        if (tid < st) { for (int o = 0; o < 4; ++o) red[o][tid] += red[o][tid + st]; }
        __syncthreads();
    }
    if (tid < 4) out[tid] = 1.f / (1.f + __expf(-(red[tid][0] + b[tid])));
}

// ---------------------------------------------------------------------------
// Orchestration
// ---------------------------------------------------------------------------
extern "C" void kernel_launch(void* const* d_in, const int* in_sizes, int n_in,
                              void* d_out, int out_size, void* d_ws, size_t ws_size,
                              hipStream_t stream)
{
    // Input leaves, insertion order of setup_inputs():
    //  0:x 1:edge_index 2:fc.w 3:fc.b
    //  4+9i: layer i -> [lin1.w, lin1.b, lin2.w, lin2.b, mlp_ln_g, mlp_ln_b, t, out_ln_g, out_ln_b]
    //  31:phi.w 32:phi.b 33:aa.w 34:aa.b 35:ab.w 36:ab.b 37:ac.w 38:ac.b 39:rho.w 40:rho.b 41:cls.w 42:cls.b
    const float* X    = (const float*)d_in[0];
    const int*   EI   = (const int*)d_in[1];
    const int*   srcp = EI;
    const int*   dstp = EI + NE;
    const float* fc_w = (const float*)d_in[2];
    const float* fc_b = (const float*)d_in[3];
    auto L = [&](int li, int j) -> const float* { return (const float*)d_in[4 + li * 9 + j]; };
    const float* phi_w = (const float*)d_in[31];
    const float* phi_b = (const float*)d_in[32];
    const float* aa_w  = (const float*)d_in[33];
    const float* aa_b  = (const float*)d_in[34];
    const float* ab_w  = (const float*)d_in[35];
    const float* ab_b  = (const float*)d_in[36];
    const float* ac_w  = (const float*)d_in[37];
    const float* ac_b  = (const float*)d_in[38];
    const float* rho_w = (const float*)d_in[39];
    const float* rho_b = (const float*)d_in[40];
    const float* cls_w = (const float*)d_in[41];
    const float* cls_b = (const float*)d_in[42];

    // Workspace carve-up. fp32 section then bf16 section. ~360 MB total.
    float* ws = (float*)d_ws;
    const long NH = (long)NN * HID;
    float* feats    = ws;                         // [NN, CATW]
    float* hp       = feats + (long)NN * CATW;    // [NN, CATW]
    float* convbuf  = hp    + (long)NN * CATW;    // 4 x [NN, HID]
    float* zmax     = convbuf;
    float* denom    = convbuf + NH;
    float* numer    = convbuf + 2 * NH;
    float* out_node = convbuf + 3 * NH;
    float* tmp1     = convbuf + 4 * NH;           // [NN, 2*HID]
    float* tmp2     = tmp1 + (long)NN * 2 * HID;  // [NN, HID]
    float* scores   = tmp2 + NH;                  // [NN]
    float* wsm      = scores + NN;                // [NN]
    float* pooled   = wsm + NN;                   // [CATW]
    float* rvec     = pooled + CATW;              // [CATW]
    float* agbuf    = convbuf;                    // reuse post-conv: [NN, CATW]
    unsigned short* a_bf = (unsigned short*)(rvec + CATW);  // [NN*CATW] bf16 A operand
    unsigned short* w_bf = a_bf + (long)NN * CATW;          // [CATW*CATW] bf16 Wt operand

    auto cvt = [&](const float* in, unsigned short* out, long n) {
        cvt_bf16_kernel<<<(int)((n / 4 + 255) / 256), 256, 0, stream>>>(in, out, n);
    };
    auto tcvt = [&](const float* W, int K, int M) {
        transpose_cvt_kernel<<<dim3(M / 32, K / 32), dim3(32, 8), 0, stream>>>(W, w_bf, K, M);
    };
    auto gemm = [&](const unsigned short* A, int lda, int ldbt, const float* bias,
                    float* C, int ldc, int nrows, int K, int M, int act) {
        dim3 grid(M / GT, (nrows + GT - 1) / GT);
        wmma_gemm_kernel<<<grid, 128, 0, stream>>>(A, lda, w_bf, ldbt, bias, C, ldc, nrows, K, act);
    };
    auto fill = [&](float* p, float v, long n) {
        fill_kernel<<<(int)((n + 255) / 256), 256, 0, stream>>>(p, v, n);
    };

    const long edgeWork = (long)NE * (HID / 4);
    const int edgeBlocks = (int)((edgeWork + 255) / 256);

    // 1) h = relu(x @ fc.w + fc.b) -> feats[:, 0:512]
    cvt(X, a_bf, (long)NN * INDIM);
    tcvt(fc_w, INDIM, HID);
    gemm(a_bf, INDIM, INDIM, fc_b, feats, CATW, NN, INDIM, HID, ACT_RELU);

    // 2) three GENConv / DeepGCN blocks
    for (int li = 0; li < 3; ++li) {
        const float* xin = feats + li * HID;          // input slice, pitch CATW
        float* cur_out   = feats + (li + 1) * HID;    // output slice, pitch CATW

        fill(zmax, -INFINITY, NH);
        fill(denom, 0.f, 2 * NH);                     // denom + numer (contiguous)

        edge_max_kernel<<<edgeBlocks, 256, 0, stream>>>(xin, CATW, srcp, dstp, L(li, 6), zmax);
        edge_sum_kernel<<<edgeBlocks, 256, 0, stream>>>(xin, CATW, srcp, dstp, L(li, 6),
                                                        zmax, denom, numer);
        node_out_kernel<<<(int)((NH + 255) / 256), 256, 0, stream>>>(xin, CATW, numer, denom, out_node);

        // MLP: lin1 -> LN+relu -> lin2
        cvt(out_node, a_bf, NH);
        tcvt(L(li, 0), HID, 2 * HID);
        gemm(a_bf, HID, HID, L(li, 1), tmp1, 2 * HID, NN, HID, 2 * HID, ACT_NONE);
        ln_kernel<<<NN, 256, 0, stream>>>(tmp1, 2 * HID, tmp1, 2 * HID,
                                          L(li, 4), L(li, 5), nullptr, 0, 2 * HID, 1);
        cvt(tmp1, a_bf, 2 * NH);
        tcvt(L(li, 2), 2 * HID, HID);
        if (li == 0) {
            gemm(a_bf, 2 * HID, 2 * HID, L(li, 3), cur_out, CATW, NN, 2 * HID, HID, ACT_NONE);
        } else {
            gemm(a_bf, 2 * HID, 2 * HID, L(li, 3), tmp2, HID, NN, 2 * HID, HID, ACT_NONE);
            ln_kernel<<<NN, 256, 0, stream>>>(tmp2, HID, cur_out, CATW,
                                              L(li, 7), L(li, 8), xin, CATW, HID, 1);
        }
    }

    // 3) hp = relu(feats @ phi + b)
    cvt(feats, a_bf, (long)NN * CATW);
    tcvt(phi_w, CATW, CATW);
    gemm(a_bf, CATW, CATW, phi_b, hp, CATW, NN, CATW, CATW, ACT_RELU);

    // 4) attention gate: agbuf = tanh(hp@aa+b);  agbuf *= sigmoid(hp@ab+b)
    cvt(hp, a_bf, (long)NN * CATW);        // hp in bf16, reused for both GEMMs
    tcvt(aa_w, CATW, CATW);
    gemm(a_bf, CATW, CATW, aa_b, agbuf, CATW, NN, CATW, CATW, ACT_TANH);
    tcvt(ab_w, CATW, CATW);
    gemm(a_bf, CATW, CATW, ab_b, agbuf, CATW, NN, CATW, CATW, ACT_SIGMUL);

    // 5) scores, node softmax, weighted pooling
    attn_score_kernel<<<NN, 256, 0, stream>>>(agbuf, ac_w, ac_b, scores);
    softmax_nodes_kernel<<<1, 1024, 0, stream>>>(scores, wsm, NN);
    fill(pooled, 0.f, CATW);
    pooled_kernel<<<dim3(CATW / 256, 40), 256, 0, stream>>>(hp, wsm, pooled, 250);

    // 6) heads
    rho_kernel<<<CATW / 256, 256, 0, stream>>>(pooled, rho_w, rho_b, rvec);
    cls_kernel<<<1, 256, 0, stream>>>(rvec, cls_w, cls_b, (float*)d_out);
}
